// MultiScaleODERNNEncoder_54417235641017
// MI455X (gfx1250) — matrix operand order repaired and checked by reading, not compiled, not run on hardware
//
#include <hip/hip_runtime.h>
#include <math.h>

// ---------------------------------------------------------------------------
// MultiScaleODERNNEncoder for MI455X (gfx1250, wave32, WMMA 16x16x32 f16,
// TDM-staged weight panels in LDS)
// B=64 T=1024 D=32 H=256 S=4 E=16 M=128 R=16 TD=19 FLAT=1024 N_EULER=5
// ---------------------------------------------------------------------------

typedef _Float16 half_t;
typedef __attribute__((ext_vector_type(16))) _Float16 v16h;
typedef __attribute__((ext_vector_type(8)))  _Float16 v8h;
typedef __attribute__((ext_vector_type(8)))  float    v8f;
typedef __attribute__((ext_vector_type(4))) unsigned int u32x4;
typedef __attribute__((ext_vector_type(8))) int          i32x8;
typedef __attribute__((ext_vector_type(4))) int          i32x4;

#define BB   64
#define TT   1024
#define DD   32
#define HH   256
#define G3H  768
#define K0P  288      // H + TD = 275 padded to multiple of 32
#define K1   512
#define FLATN 1024

__device__ __forceinline__ float sigmoidf_(float x) { return 1.f / (1.f + expf(-x)); }
__device__ __forceinline__ float geluf_(float x)    { return 0.5f * x * (1.f + erff(x * 0.70710678118f)); }
__device__ __forceinline__ float tfreq_(int k)      { return 6.28318530718f * powf(1000.f, -(float)k / 7.f); }

// --- WMMA fragment loaders (layouts per CDNA5 ISA 7.12.2, wave32) -----------
// A 16x32 f16 (row-major, lda elems): lane m=lane&15 holds K=kb..kb+7 and
// kb+16..kb+23, kb = (lane&16)?8:0
__device__ __forceinline__ v16h load_a16x32(const half_t* A, int lda, int lane) {
  int m  = lane & 15;
  int kb = (lane & 16) ? 8 : 0;
  const half_t* p = A + (size_t)m * lda + kb;
  v8h lo = *(const v8h*)(p);
  v8h hi = *(const v8h*)(p + 16);
  v16h a;
#pragma unroll
  for (int i = 0; i < 8; ++i) { a[i] = lo[i]; a[i + 8] = hi[i]; }
  return a;
}

// B 32x16 f16 from W (N x K row-major): lane holds column n=lane&15,
// K = kb..kb+7 and kb+16..kb+23, kb = (lane&16)?8:0  => two 16B loads per lane.
__device__ __forceinline__ v16h load_b32x16(const half_t* W, int ldw, int lane) {
  int n  = lane & 15;
  int kb = (lane & 16) ? 8 : 0;
  const half_t* p = W + (size_t)n * ldw + kb;
  v8h lo = *(const v8h*)(p);
  v8h hi = *(const v8h*)(p + 16);
  v16h b;
#pragma unroll
  for (int i = 0; i < 8; ++i) { b[i] = lo[i]; b[i + 8] = hi[i]; }
  return b;
}

// ---------------------------------------------------------------------------
// WMMA GEMM: C(M,N) f32 = A(M,K) f16 @ W(N,K)^T f16 (+ bias)
// Block = 8 waves sharing one weight panel (16 rows x K f16, <=32KB) staged
// into LDS by the Tensor Data Mover; wave w computes row-tile tmb+w.
// TDM path is unconditional in the device pass: compile failure here means the
// builtin signature differs -- we want that signal, not a silent fallback.
// ---------------------------------------------------------------------------
__global__ void __launch_bounds__(256)
gemm_wmma_f16(const half_t* __restrict__ A,
              const half_t* __restrict__ W,
              const float*  __restrict__ bias,
              float* __restrict__ C,
              int Mrows, int N, int K) {
  __shared__ half_t wpanel[16 * 1024];
  int lane = threadIdx.x & 31;
  int wave = threadIdx.x >> 5;          // 0..7
  int tiles_n = N >> 4;
  int tiles_m = Mrows >> 4;
  int tn  = blockIdx.x % tiles_n;
  int tmb = (blockIdx.x / tiles_n) * 8;

  const half_t* Wp = W + (size_t)tn * 16 * K;   // contiguous 16*K f16 panel

#if defined(__HIP_DEVICE_COMPILE__)
  if (threadIdx.x < 32) {               // wave 0 issues one TDM descriptor
    unsigned elems = 16u * (unsigned)K;                 // <= 16384 (fits tile_dim0)
    unsigned lds   = (unsigned)(size_t)(void*)wpanel;   // LDS aperture: addr[31:0]
    unsigned long long ga = (unsigned long long)(size_t)Wp;
    u32x4 g0 = { 1u,                                    // count=1, user mode
                 lds,                                   // lds_addr
                 (unsigned)(ga & 0xffffffffu),          // global_addr[31:0]
                 (unsigned)((ga >> 32) & 0x01ffffffu) | (2u << 30) }; // addr hi | type=2
    i32x8 g1 = { (int)(1u << 16),                       // data_size=1 (2 bytes)
                 (int)(elems << 16),                    // tensor_dim0[15:0]
                 (int)((elems >> 16) & 0xffffu) | (1 << 16), // dim0 hi | tensor_dim1=1
                 (int)(elems << 16),                    // tile_dim0 (1-D tile)
                 0,                                     // tile_dim1=0, tile_dim2=0
                 (int)elems,                            // tensor_dim0_stride lo
                 0, 0 };
    i32x4 gz = { 0, 0, 0, 0 };
#if __clang_major__ >= 23
    i32x8 gz8 = { 0, 0, 0, 0, 0, 0, 0, 0 };
    __builtin_amdgcn_tensor_load_to_lds(g0, g1, gz, gz, gz8, 0);
#else
    __builtin_amdgcn_tensor_load_to_lds(g0, g1, gz, gz, 0);
#endif
    __builtin_amdgcn_s_wait_tensorcnt(0);
  }
  __syncthreads();
#else
  // host-pass parse fallback (never executed)
  {
    const v8h* src = (const v8h*)Wp;
    v8h* dst = (v8h*)wpanel;
    for (int i = threadIdx.x; i < 2 * K; i += 256) dst[i] = src[i];
    __syncthreads();
  }
#endif

  int tm = tmb + wave;
  if (tm < tiles_m) {
    const half_t* Ap = A + (size_t)tm * 16 * K;
    v8f acc = {};
    for (int k = 0; k < K; k += 32) {
      v16h a = load_a16x32(Ap + k, K, lane);
      v16h b = load_b32x16(wpanel + k, K, lane);   // ds_load path
      acc = __builtin_amdgcn_wmma_f32_16x16x32_f16(false, a, false, b, (short)0, acc, false, false);
    }
    int col = tn * 16 + (lane & 15);
    float bv = bias ? bias[col] : 0.f;
    int mb = tm * 16 + ((lane & 16) ? 8 : 0);
#pragma unroll
    for (int j = 0; j < 8; ++j)
      C[(size_t)(mb + j) * N + col] = acc[j] + bv;
  }
}

// ---------------------------------------------------------------------------
// Prep: obs proj + layernorm + gelu + time features -> gi_f16 (B*T, 288)
// ---------------------------------------------------------------------------
__global__ void prep_kernel(const float* __restrict__ times,
                            const float* __restrict__ values,
                            const float* __restrict__ mask,
                            const float* __restrict__ inp_W,
                            const float* __restrict__ inp_b,
                            const float* __restrict__ ln_g,
                            const float* __restrict__ ln_b,
                            half_t* __restrict__ gi,
                            float* __restrict__ obsmask) {
  int bt = blockIdx.x;                 // 0..B*T-1
  int t  = bt & (TT - 1);
  int tid = threadIdx.x;               // 256
  __shared__ float obs[2 * DD];
  __shared__ float red[256];

  if (tid < DD) {
    float v  = values[(size_t)bt * DD + tid];
    float mk = mask[(size_t)bt * DD + tid];
    obs[tid]      = v * mk;
    obs[DD + tid] = mk;
  }
  __syncthreads();

  float acc = inp_b[tid];
  const float* w = inp_W + (size_t)tid * (2 * DD);
#pragma unroll 8
  for (int k = 0; k < 2 * DD; ++k) acc += obs[k] * w[k];

  red[tid] = acc; __syncthreads();
  for (int o = 128; o > 0; o >>= 1) { if (tid < o) red[tid] += red[tid + o]; __syncthreads(); }
  float mu = red[0] / 256.f; __syncthreads();
  float d = acc - mu;
  red[tid] = d * d; __syncthreads();
  for (int o = 128; o > 0; o >>= 1) { if (tid < o) red[tid] += red[tid + o]; __syncthreads(); }
  float var = red[0] / 256.f;

  float x = d * rsqrtf(var + 1e-5f) * ln_g[tid] + ln_b[tid];
  half_t* grow = gi + (size_t)bt * K0P;
  grow[tid] = (half_t)geluf_(x);

  if (tid < 32) {
    float val = 0.f;
    if (tid < 19) {
      float tv = times[bt];
      float dt = (t == 0) ? 0.f : tv - times[bt - 1];
      if (tid < 8)       val = sinf(tv * tfreq_(tid));
      else if (tid < 16) val = cosf(tv * tfreq_(tid - 8));
      else if (tid == 16) val = 1.f;
      else if (tid == 17) val = dt;
      else                val = dt * dt;
    }
    grow[HH + tid] = (half_t)val;     // cols 256..287 (275..287 zero pad)
  }
  if (tid == 0) {
    float s = 0.f;
    for (int k = 0; k < DD; ++k) s += mask[(size_t)bt * DD + k];
    obsmask[bt] = (s > 0.f) ? 1.f : 0.f;
  }
}

// ---------------------------------------------------------------------------
// GRU scan (persistent): gridDim.x = 8  -> (dir, mblock of 16 batch rows)
// h in LDS (f32 master + f16 WMMA copy); Whh L2-resident; 24 WMMA/step/wave
// ---------------------------------------------------------------------------
__global__ void __launch_bounds__(512)
gru_scan(const float* __restrict__ gatesA,   // dir0 pre (B*T,768), bih included
         const float* __restrict__ gatesB,   // dir1 pre
         const half_t* __restrict__ Whh,     // (2,768,256) f16
         const float* __restrict__ bhh,      // (2,768)
         half_t* __restrict__ out16, int ld16,
         float*  __restrict__ out32, int ld32) {
  int dir   = blockIdx.x & 1;
  int mbase = (blockIdx.x >> 1) * 16;
  const float*  gates = dir ? gatesB : gatesA;
  const half_t* Wd = Whh + (size_t)dir * G3H * HH;
  const float*  bd = bhh + dir * G3H;

  __shared__ half_t hh[16][HH];   // 8 KB  (WMMA operand copy)
  __shared__ float  hf[16][HH];   // 16 KB (master state)
  for (int i = threadIdx.x; i < 16 * HH; i += blockDim.x) {
    (&hh[0][0])[i] = (half_t)0.f;
    (&hf[0][0])[i] = 0.f;
  }
  __syncthreads();

  int lane  = threadIdx.x & 31;
  int wave  = threadIdx.x >> 5;        // 0..15 -> column tile of H
  int cbase = wave * 16;
  int n     = lane & 15;
  int c     = cbase + n;
  int mlo   = (lane & 16) ? 8 : 0;
  float br = bd[c], bz = bd[HH + c], bn = bd[2 * HH + c];

  const half_t* Wr = Wd + (size_t)(cbase) * HH;
  const half_t* Wz = Wd + (size_t)(HH  + cbase) * HH;
  const half_t* Wn = Wd + (size_t)(2*HH + cbase) * HH;

  for (int t = 0; t < TT; ++t) {
    int tt = dir ? (TT - 1 - t) : t;

    // prefetch next step's precomputed gate rows toward this WGP
    if (t + 1 < TT) {
      int tn2 = dir ? (TT - 2 - t) : (t + 1);
      size_t gp = ((size_t)(mbase + mlo) * TT + tn2) * G3H + c;
      __builtin_prefetch(&gates[gp], 0, 0);
    }

    v8f ar = {}, az = {}, an = {};
#pragma unroll
    for (int k = 0; k < HH; k += 32) {
      v16h a  = load_a16x32(&hh[0][0] + k, HH, lane);
      v16h b0 = load_b32x16(Wr + k, HH, lane);
      v16h b1 = load_b32x16(Wz + k, HH, lane);
      v16h b2 = load_b32x16(Wn + k, HH, lane);
      ar = __builtin_amdgcn_wmma_f32_16x16x32_f16(false, a, false, b0, (short)0, ar, false, false);
      az = __builtin_amdgcn_wmma_f32_16x16x32_f16(false, a, false, b1, (short)0, az, false, false);
      an = __builtin_amdgcn_wmma_f32_16x16x32_f16(false, a, false, b2, (short)0, an, false, false);
    }
    __syncthreads();   // all hh reads complete before update

#pragma unroll
    for (int j = 0; j < 8; ++j) {
      int m = mlo + j;
      int b = mbase + m;
      size_t gidx = ((size_t)b * TT + tt) * G3H;
      float gir = gates[gidx + c];
      float giz = gates[gidx + HH + c];
      float gin = gates[gidx + 2 * HH + c];
      float hold = hf[m][c];
      float r  = sigmoidf_(gir + ar[j] + br);
      float zg = sigmoidf_(giz + az[j] + bz);
      float nn = tanhf(gin + r * (an[j] + bn));
      float hnew = (1.f - zg) * nn + zg * hold;
      hf[m][c] = hnew;
      hh[m][c] = (half_t)hnew;
      size_t orow = (size_t)b * TT + tt;
      if (out16) out16[orow * ld16 + dir * HH + c] = (half_t)hnew;
      if (out32) out32[orow * ld32 + dir * HH + c] = hnew;
    }
    __syncthreads();   // updates visible before next step's reads
  }
}

// ---------------------------------------------------------------------------
// Pooling: pooled(b) = [ g2[b,T-1,:] (512) , masked-mean_t g2 (512) ]
// ---------------------------------------------------------------------------
__global__ void pool_kernel(const float* __restrict__ g2,
                            const float* __restrict__ obsmask,
                            float* __restrict__ pooled) {
  int b = blockIdx.x, tid = threadIdx.x;   // 256 threads, dims tid & tid+256
  float s0 = 0.f, s1 = 0.f, cnt = 0.f;
  for (int t = 0; t < TT; ++t) {
    float om = obsmask[b * TT + t];
    size_t base = ((size_t)b * TT + t) * 512;
    s0 += g2[base + tid] * om;
    s1 += g2[base + tid + 256] * om;
    cnt += om;
  }
  float den = fmaxf(cnt, 1.f);
  size_t lastb = ((size_t)b * TT + (TT - 1)) * 512;
  size_t ob = (size_t)b * FLATN;
  pooled[ob + tid]             = g2[lastb + tid];
  pooled[ob + tid + 256]       = g2[lastb + tid + 256];
  pooled[ob + 512 + tid]       = s0 / den;
  pooled[ob + 512 + tid + 256] = s1 / den;
}

// ---------------------------------------------------------------------------
// LayerNorm over C=1024 per row (optional elementwise add; f32 and/or f16 out)
// ---------------------------------------------------------------------------
__global__ void ln_kernel(const float* __restrict__ x, const float* __restrict__ addx,
                          const float* __restrict__ g, const float* __restrict__ b,
                          float* __restrict__ y32, half_t* __restrict__ y16) {
  int row = blockIdx.x, tid = threadIdx.x;   // 256 threads x 4 elems
  __shared__ float red[256];
  const size_t base = (size_t)row * FLATN;
  float v[4]; float s = 0.f;
#pragma unroll
  for (int i = 0; i < 4; ++i) {
    int cc = tid + i * 256;
    float t = x[base + cc];
    if (addx) t += addx[base + cc];
    v[i] = t; s += t;
  }
  red[tid] = s; __syncthreads();
  for (int o = 128; o > 0; o >>= 1) { if (tid < o) red[tid] += red[tid + o]; __syncthreads(); }
  float mu = red[0] / (float)FLATN; __syncthreads();
  float s2 = 0.f;
#pragma unroll
  for (int i = 0; i < 4; ++i) { float d = v[i] - mu; s2 += d * d; }
  red[tid] = s2; __syncthreads();
  for (int o = 128; o > 0; o >>= 1) { if (tid < o) red[tid] += red[tid + o]; __syncthreads(); }
  float inv = rsqrtf(red[0] / (float)FLATN + 1e-5f);
#pragma unroll
  for (int i = 0; i < 4; ++i) {
    int cc = tid + i * 256;
    float o = (v[i] - mu) * inv * g[cc] + b[cc];
    if (y32) y32[base + cc] = o;
    if (y16) y16[base + cc] = (half_t)o;
  }
}

// ---------------------------------------------------------------------------
// small utilities
// ---------------------------------------------------------------------------
__global__ void f32_to_f16_pad(const float* __restrict__ src, half_t* __restrict__ dst,
                               int N, int Kin, int Kpad) {
  size_t tot = (size_t)N * Kpad;
  for (size_t i = (size_t)blockIdx.x * blockDim.x + threadIdx.x; i < tot;
       i += (size_t)gridDim.x * blockDim.x) {
    size_t r = i / Kpad; int k = (int)(i % Kpad);
    dst[i] = (k < Kin) ? (half_t)src[r * Kin + k] : (half_t)0.f;
  }
}

__global__ void copy_f32(const float* __restrict__ a, float* __restrict__ b, int n) {
  for (int i = blockIdx.x * blockDim.x + threadIdx.x; i < n; i += gridDim.x * blockDim.x)
    b[i] = a[i];
}

// ---------------------------------------------------------------------------
// ODE: cu(b,r) = zf @ coup_U  (tiny)
// ---------------------------------------------------------------------------
__global__ void ode_cu_kernel(const float* __restrict__ z, const float* __restrict__ coupU,
                              float* __restrict__ cu) {
  int b = blockIdx.x, r = threadIdx.x;
  if (r < 16) {
    float s = 0.f;
    for (int k = 0; k < FLATN; ++k) s += z[(size_t)b * FLATN + k] * coupU[(size_t)k * 16 + r];
    cu[b * 16 + r] = s;
  }
}

// one Euler step for (b,s); in-place on z_ev
__global__ void ode_step_kernel(float* __restrict__ z,
                                const float* __restrict__ scale_emb,
                                const float* __restrict__ W1, const float* __restrict__ b1,
                                const float* __restrict__ W2, const float* __restrict__ b2,
                                const float* __restrict__ tau_raw,
                                const float* __restrict__ cu, const float* __restrict__ coupV,
                                float tcur) {
  int bs = blockIdx.x; int b = bs >> 2; int s = bs & 3;
  int tid = threadIdx.x;    // 256
  __shared__ float inp[292];
  __shared__ float hid[128];

  inp[tid] = z[(size_t)b * FLATN + s * 256 + tid];
  if (tid < 16)       inp[256 + tid] = scale_emb[s * 16 + tid];
  else if (tid < 24)  inp[272 + (tid - 16)] = sinf(tcur * tfreq_(tid - 16));
  else if (tid < 32)  inp[280 + (tid - 24)] = cosf(tcur * tfreq_(tid - 24));
  else if (tid == 32) { inp[288] = 1.f; inp[289] = 0.f; inp[290] = 0.f; }
  __syncthreads();

  if (tid < 128) {
    float a = b1[tid];
    const float* w = W1 + (size_t)tid * 291;
    for (int k = 0; k < 291; ++k) a += inp[k] * w[k];
    hid[tid] = geluf_(a);
  }
  __syncthreads();

  float m = b2[tid];
  const float* w2 = W2 + (size_t)tid * 128;
#pragma unroll 8
  for (int k = 0; k < 128; ++k) m += hid[k] * w2[k];

  float tau = 0.01f + (100.f - 0.01f) * sigmoidf_(tau_raw[s]);
  float cp = 0.f;
  const float* cvr = coupV + (size_t)(s * 256 + tid) * 16;
#pragma unroll
  for (int r = 0; r < 16; ++r) cp += cu[b * 16 + r] * cvr[r];

  float zc = inp[tid];
  float dz = (m - zc) / tau + 0.1f * cp;
  z[(size_t)b * FLATN + s * 256 + tid] = zc + 0.2f * dz;   // step = 1/5
}

// ---------------------------------------------------------------------------
// host-side orchestration
// ---------------------------------------------------------------------------
extern "C" void kernel_launch(void* const* d_in, const int* in_sizes, int n_in,
                              void* d_out, int out_size, void* d_ws, size_t ws_size,
                              hipStream_t stream) {
  (void)in_sizes; (void)n_in; (void)out_size; (void)ws_size;
  const float* times    = (const float*)d_in[0];
  const float* values   = (const float*)d_in[1];
  const float* mask     = (const float*)d_in[2];
  const float* inp_W    = (const float*)d_in[3];
  const float* inp_b    = (const float*)d_in[4];
  const float* inp_ln_g = (const float*)d_in[5];
  const float* inp_ln_b = (const float*)d_in[6];
  const float* g0Wih    = (const float*)d_in[7];
  const float* g0Whh    = (const float*)d_in[8];
  const float* g0bih    = (const float*)d_in[9];
  const float* g0bhh    = (const float*)d_in[10];
  const float* g1Wih    = (const float*)d_in[11];
  const float* g1Whh    = (const float*)d_in[12];
  const float* g1bih    = (const float*)d_in[13];
  const float* g1bhh    = (const float*)d_in[14];
  const float* ms_W     = (const float*)d_in[15];
  const float* ms_b     = (const float*)d_in[16];
  const float* ms_ln_g  = (const float*)d_in[17];
  const float* ms_ln_b  = (const float*)d_in[18];
  const float* scale_emb= (const float*)d_in[19];
  const float* ode_W1   = (const float*)d_in[20];
  const float* ode_b1   = (const float*)d_in[21];
  const float* ode_W2   = (const float*)d_in[22];
  const float* ode_b2   = (const float*)d_in[23];
  const float* tau_raw  = (const float*)d_in[24];
  const float* coup_U   = (const float*)d_in[25];
  const float* coup_V   = (const float*)d_in[26];
  const float* out_ln_g = (const float*)d_in[27];
  const float* out_ln_b = (const float*)d_in[28];
  const float* out_W    = (const float*)d_in[29];
  const float* out_b    = (const float*)d_in[30];
  float* out = (float*)d_out;

  char* ws = (char*)d_ws;
  size_t off = 0;
  auto alloc = [&](size_t bytes) -> char* {
    char* p = ws + off; off = (off + bytes + 255) & ~(size_t)255; return p;
  };
  const size_t BT = (size_t)BB * TT;
  half_t* gi16    = (half_t*)alloc(BT * K0P * 2);
  half_t* w0ih16  = (half_t*)alloc((size_t)2 * G3H * K0P * 2);
  half_t* w0hh16  = (half_t*)alloc((size_t)2 * G3H * HH * 2);
  half_t* w1ih16  = (half_t*)alloc((size_t)2 * G3H * K1 * 2);
  half_t* w1hh16  = (half_t*)alloc((size_t)2 * G3H * HH * 2);
  float*  gates   = (float*) alloc((size_t)2 * BT * G3H * 4);  // reused for L0 & L1
  half_t* g1_16   = (half_t*)alloc(BT * 512 * 2);
  float*  g2_32   = (float*) alloc(BT * 512 * 4);
  float*  obsm    = (float*) alloc(BT * 4);
  float*  pooled  = (float*) alloc((size_t)BB * FLATN * 4);
  half_t* pooled16= (half_t*)alloc((size_t)BB * FLATN * 2);
  half_t* msW16   = (half_t*)alloc((size_t)FLATN * FLATN * 2);
  half_t* outW16  = (half_t*)alloc((size_t)FLATN * FLATN * 2);
  float*  mslin   = (float*) alloc((size_t)BB * FLATN * 4);
  float*  z0      = (float*) alloc((size_t)BB * FLATN * 4);
  float*  zev     = (float*) alloc((size_t)BB * FLATN * 4);
  float*  cu      = (float*) alloc((size_t)BB * 16 * 4);
  half_t* ln16    = (half_t*)alloc((size_t)BB * FLATN * 2);

  // --- weight conversions to f16 (padded where needed) ---
  f32_to_f16_pad<<<2048, 256, 0, stream>>>(g0Wih, w0ih16, 2 * G3H, 275, K0P);
  f32_to_f16_pad<<<2048, 256, 0, stream>>>(g0Whh, w0hh16, 2 * G3H, HH, HH);
  f32_to_f16_pad<<<2048, 256, 0, stream>>>(g1Wih, w1ih16, 2 * G3H, K1, K1);
  f32_to_f16_pad<<<2048, 256, 0, stream>>>(g1Whh, w1hh16, 2 * G3H, HH, HH);
  f32_to_f16_pad<<<2048, 256, 0, stream>>>(ms_W,  msW16,  FLATN, FLATN, FLATN);
  f32_to_f16_pad<<<2048, 256, 0, stream>>>(out_W, outW16, FLATN, FLATN, FLATN);

  // --- prep: input proj + LN + gelu + time features ---
  prep_kernel<<<(int)BT, 256, 0, stream>>>(times, values, mask, inp_W, inp_b,
                                           inp_ln_g, inp_ln_b, gi16, obsm);

  auto gemm_blocks = [](int tiles_m, int tiles_n) {
    return ((tiles_m + 7) / 8) * tiles_n;
  };

  // --- layer0 input GEMMs (WMMA + TDM-staged weight panels), both dirs ---
  {
    int blocks = gemm_blocks((int)(BT / 16), G3H / 16);
    gemm_wmma_f16<<<blocks, 256, 0, stream>>>(gi16, w0ih16, g0bih, gates,
                                              (int)BT, G3H, K0P);
    gemm_wmma_f16<<<blocks, 256, 0, stream>>>(gi16, w0ih16 + (size_t)G3H * K0P,
                                              g0bih + G3H, gates + BT * G3H,
                                              (int)BT, G3H, K0P);
  }
  // --- layer0 scan (8 persistent blocks: 2 dirs x 4 batch slices) ---
  gru_scan<<<8, 512, 0, stream>>>(gates, gates + BT * G3H, w0hh16, g0bhh,
                                  g1_16, 512, (float*)nullptr, 0);

  // --- layer1 input GEMMs ---
  {
    int blocks = gemm_blocks((int)(BT / 16), G3H / 16);
    gemm_wmma_f16<<<blocks, 256, 0, stream>>>(g1_16, w1ih16, g1bih, gates,
                                              (int)BT, G3H, K1);
    gemm_wmma_f16<<<blocks, 256, 0, stream>>>(g1_16, w1ih16 + (size_t)G3H * K1,
                                              g1bih + G3H, gates + BT * G3H,
                                              (int)BT, G3H, K1);
  }
  // --- layer1 scan -> g2 (f32) ---
  gru_scan<<<8, 512, 0, stream>>>(gates, gates + BT * G3H, w1hh16, g1bhh,
                                  (half_t*)nullptr, 0, g2_32, 512);

  // --- pooling, ms projection (WMMA), layernorm -> z0 ---
  pool_kernel<<<BB, 256, 0, stream>>>(g2_32, obsm, pooled);
  f32_to_f16_pad<<<256, 256, 0, stream>>>(pooled, pooled16, BB, FLATN, FLATN);
  {
    int blocks = gemm_blocks(BB / 16, FLATN / 16);
    gemm_wmma_f16<<<blocks, 256, 0, stream>>>(pooled16, msW16, ms_b, mslin,
                                              BB, FLATN, FLATN);
  }
  ln_kernel<<<BB, 256, 0, stream>>>(mslin, (const float*)nullptr, ms_ln_g, ms_ln_b,
                                    z0, (half_t*)nullptr);
  copy_f32<<<256, 256, 0, stream>>>(z0, zev, BB * FLATN);

  // --- 5 Euler ODE steps ---
  for (int i = 0; i < 5; ++i) {
    ode_cu_kernel<<<BB, 64, 0, stream>>>(zev, coup_U, cu);
    ode_step_kernel<<<BB * 4, 256, 0, stream>>>(zev, scale_emb, ode_W1, ode_b1,
                                                ode_W2, ode_b2, tau_raw, cu, coup_V,
                                                (float)i * 0.2f);
  }

  // --- final: LN(z_ev + z0) then output GEMM (WMMA) + bias -> d_out ---
  ln_kernel<<<BB, 256, 0, stream>>>(zev, z0, out_ln_g, out_ln_b,
                                    (float*)nullptr, ln16);
  {
    int blocks = gemm_blocks(BB / 16, FLATN / 16);
    gemm_wmma_f16<<<blocks, 256, 0, stream>>>(ln16, outW16, out_b, out,
                                              BB, FLATN, FLATN);
  }
}
